// CLGNN_GCN_87196426043919
// MI455X (gfx1250) — matrix operand order repaired and checked by reading, not compiled, not run on hardware
//
#include <hip/hip_runtime.h>
#include <hip/hip_bf16.h>

typedef __attribute__((ext_vector_type(16))) _Float16 v16h;
typedef __attribute__((ext_vector_type(8)))  float    v8f;

#define NNODES   100000
#define FIN      256
#define HID1     128
#define HID2     64
#define DEC1     128
#define OFFS     10367

// ------------------------- utility kernels -------------------------

__global__ void k_fill(float* __restrict__ p, float v, int n) {
  int i = blockIdx.x * blockDim.x + threadIdx.x;
  if (i < n) p[i] = v;
}

__global__ void k_deg_edges(const int* __restrict__ dst, float* __restrict__ deg, int E) {
  int e = blockIdx.x * blockDim.x + threadIdx.x;
  if (e < E) atomicAdd(&deg[dst[e]], 1.0f);
}

__global__ void k_rsqrt_inplace(float* __restrict__ p, int n) {
  int i = blockIdx.x * blockDim.x + threadIdx.x;
  if (i < n) p[i] = rsqrtf(p[i]);
}

// transpose+convert weights: Wt[n][k] (f16, col-major) from W[k][n] (f32, row-major)
template<int K, int NCOL>
__global__ void k_wt(const float* __restrict__ W, _Float16* __restrict__ Wt) {
  int t = blockIdx.x * blockDim.x + threadIdx.x;
  if (t >= K * NCOL) return;
  int n = t / K, k = t - n * K;
  Wt[t] = (_Float16)W[(size_t)k * NCOL + n];
}

// ------------------------- WMMA GEMM: Y[M,NCOL] = X[M,K] @ W[K,NCOL] -------------------------
// One wave computes a full 16 x NCOL row-block: the A fragment is loaded/converted once per
// k-step and reused by NCOL/16 WMMAs. Wt is the weight pre-transposed to col-major f16 so each
// B fragment (16 consecutive K values at one column) is a single contiguous 32B load.
// Fragment layouts used here (same contiguous k order for A and B):
//   A (16x32 f16): lane L -> row = L&15, half = L>>4; elems 0..7 = K k0+half*16+i,
//                  elems 8..15 = K k0+half*16+8+i
//   B (32x16 f16): lane L -> col = L&15; elem j = K k0+half*16+j (contiguous!)
//   C/D (16x16 f32): VGPR r -> row r+8*half, col L&15
// M is a multiple of 16 here (100000, 8192); early-exit is wave-uniform so EXEC stays all-1s.
template<int K, int NCOL>
__global__ void k_gemm_wmma(const float* __restrict__ X,
                            const _Float16* __restrict__ Wt,   // [NCOL][K] f16
                            float* __restrict__ Y, int M) {
  constexpr int NT = NCOL / 16;
  const int lane = threadIdx.x & 31;
  const int wave = threadIdx.x >> 5;
  const int tm   = blockIdx.x * (blockDim.x >> 5) + wave;
  if (tm * 16 >= M) return;              // wave-uniform
  const int half = lane >> 4;
  const int l15  = lane & 15;
  const int m    = tm * 16 + l15;

  v8f acc[NT];
#pragma unroll
  for (int t = 0; t < NT; ++t) acc[t] = (v8f){};

  for (int k0 = 0; k0 < K; k0 += 32) {
    const float* xr = X + (size_t)m * K + k0;
    v16h a;
#pragma unroll
    for (int i = 0; i < 8; ++i) a[i]     = (_Float16)xr[half * 16 + i];
#pragma unroll
    for (int i = 0; i < 8; ++i) a[8 + i] = (_Float16)xr[half * 16 + 8 + i];

#pragma unroll
    for (int t = 0; t < NT; ++t) {
      v16h b = *(const v16h*)(Wt + (size_t)(t * 16 + l15) * K + k0 + half * 16);
      acc[t] = __builtin_amdgcn_wmma_f32_16x16x32_f16(
          false, a, false, b, (short)0, acc[t], false, false);
    }
  }

#pragma unroll
  for (int t = 0; t < NT; ++t) {
    float* yr = Y + (size_t)(tm * 16 + half * 8) * NCOL + t * 16 + l15;
#pragma unroll
    for (int r = 0; r < 8; ++r) yr[(size_t)r * NCOL] = acc[t][r];
  }
}

// ------------------------- edge aggregation: agg[dst] += h[src] * dinv[src]*dinv[dst] -----
// One wave per edge; lanes stride the feature dimension. Atomics land in L2 (agg fits).
template<int F>
__global__ void k_agg_edges(const int* __restrict__ src, const int* __restrict__ dst,
                            const float* __restrict__ dinv,
                            const float* __restrict__ h, float* __restrict__ agg, int E) {
  const int lane = threadIdx.x & 31;
  const int e = (blockIdx.x * blockDim.x + threadIdx.x) >> 5;
  if (e >= E) return;
  const int s = src[e], d = dst[e];
  const float nrm = dinv[s] * dinv[d];
  const float* hs = h + (size_t)s * F;
  float* ad = agg + (size_t)d * F;
#pragma unroll
  for (int k = 0; k < F / 32; ++k)
    atomicAdd(&ad[lane + 32 * k], hs[lane + 32 * k] * nrm);
}

// self-loop term + bias + PReLU:  out = prelu(agg + h*dinv^2 + bias)
template<int F>
__global__ void k_layer_post(const float* __restrict__ agg, const float* __restrict__ h,
                             const float* __restrict__ dinv, const float* __restrict__ bias,
                             const float* __restrict__ slope, float* __restrict__ out, int Nn) {
  int i = blockIdx.x * blockDim.x + threadIdx.x;
  int node = i / F, f = i - node * F;
  if (node >= Nn) return;
  float di = dinv[node];
  float v = agg[i] + h[i] * di * di + bias[f];
  out[i] = (v >= 0.f) ? v : slope[f] * v;
}

// ------------------------- summary / discriminator -------------------------

__global__ void k_colsum(const float* __restrict__ x, float* __restrict__ csum, int Nn) {
  const int CHUNK = 512;
  int j  = threadIdx.x;                  // 64 threads = 64 columns
  int r0 = blockIdx.x * CHUNK;
  int r1 = r0 + CHUNK; if (r1 > Nn) r1 = Nn;
  float s = 0.f;
  for (int r = r0; r < r1; ++r) s += x[(size_t)r * 64 + j];
  atomicAdd(&csum[j], s);
}

__global__ void k_summary(const float* __restrict__ csum,
                          const float* __restrict__ mlp1_W, const float* __restrict__ mlp1_b,
                          const float* __restrict__ disc_W, float* __restrict__ Wc) {
  __shared__ float c0[64], c1[64];
  int j = threadIdx.x;
  float m = csum[j] / (float)NNODES;
  c0[j] = 1.f / (1.f + __expf(-m));
  __syncthreads();
  float s = mlp1_b[j];
  for (int k = 0; k < 64; ++k) s += c0[k] * mlp1_W[k * 64 + j];
  c1[j] = s;
  __syncthreads();
  float w = 0.f;
  for (int k = 0; k < 64; ++k) w += disc_W[j * 64 + k] * c1[k];
  Wc[j] = w;
}

__global__ void k_disc(const float* __restrict__ x2o, const float* __restrict__ x2a,
                       const float* __restrict__ Wc, const float* __restrict__ disc_b,
                       float* __restrict__ ret, int Nn) {
  __shared__ float wc[64];
  if (threadIdx.x < 64) wc[threadIdx.x] = Wc[threadIdx.x];
  __syncthreads();
  int i = blockIdx.x * blockDim.x + threadIdx.x;
  if (i >= Nn) return;
  const float bb = disc_b[0];
  const float* r1 = x2o + (size_t)i * 64;
  const float* r2 = x2a + (size_t)i * 64;
  float s1 = 0.f, s2 = 0.f;
#pragma unroll 8
  for (int k = 0; k < 64; ++k) { s1 += r1[k] * wc[k]; s2 += r2[k] * wc[k]; }
  ret[2 * i]     = s1 + bb;
  ret[2 * i + 1] = s2 + bb;
}

// ------------------------- decoder -------------------------

__global__ void k_feat(const float* __restrict__ x2o, const int* __restrict__ idx0,
                       const int* __restrict__ idx1, float* __restrict__ feat, int B) {
  int t = blockIdx.x * blockDim.x + threadIdx.x;
  int b = t >> 6, j = t & 63;
  if (b >= B) return;
  float e1 = x2o[(size_t)idx0[b] * 64 + j];
  float e2 = x2o[((size_t)idx1[b] + OFFS) * 64 + j];
  float* fb = feat + (size_t)b * 256;
  fb[j]       = e1 + e2;
  fb[64 + j]  = e1 * e2;
  fb[128 + j] = e1;
  fb[192 + j] = e2;
}

__global__ void k_bias_relu(float* __restrict__ p, const float* __restrict__ bias, int n, int F) {
  int i = blockIdx.x * blockDim.x + threadIdx.x;
  if (i >= n) return;
  float v = p[i] + bias[i % F];
  p[i] = (v > 0.f) ? v : 0.f;
}

__global__ void k_dec2(const float* __restrict__ log1, const float* __restrict__ w,
                       const float* __restrict__ b, float* __restrict__ out, int B) {
  int i = blockIdx.x * blockDim.x + threadIdx.x;
  if (i >= B) return;
  const float* r = log1 + (size_t)i * 128;
  float s = b[0];
#pragma unroll 8
  for (int k = 0; k < 128; ++k) s += r[k] * w[k];
  out[i] = s;
}

// ------------------------- driver -------------------------

extern "C" void kernel_launch(void* const* d_in, const int* in_sizes, int n_in,
                              void* d_out, int out_size, void* d_ws, size_t ws_size,
                              hipStream_t stream) {
  const float* x_o    = (const float*)d_in[0];
  const float* x_a    = (const float*)d_in[1];
  const int*   eidx   = (const int*)  d_in[2];
  const int*   idx    = (const int*)  d_in[3];
  const float* W1     = (const float*)d_in[4];
  const float* b1     = (const float*)d_in[5];
  const float* a1     = (const float*)d_in[6];
  const float* W2     = (const float*)d_in[7];
  const float* b2     = (const float*)d_in[8];
  const float* a2     = (const float*)d_in[9];
  const float* mlp1_W = (const float*)d_in[10];
  const float* mlp1_b = (const float*)d_in[11];
  const float* disc_W = (const float*)d_in[12];
  const float* disc_b = (const float*)d_in[13];
  const float* dec1_W = (const float*)d_in[14];
  const float* dec1_b = (const float*)d_in[15];
  const float* dec2_W = (const float*)d_in[16];
  const float* dec2_b = (const float*)d_in[17];

  const int E  = in_sizes[2] / 2;
  const int B  = in_sizes[3] / 2;
  const int Nn = NNODES;

  // workspace (floats): dinv[Npad] | csum[64] | Wc[64] | pad | Wt region (40960 floats) |
  //                     R1[N*128] | R2[N*128]
  float* ws   = (float*)d_ws;
  const size_t Npad = 100096;
  float* dinv = ws;
  float* csum = ws + Npad;
  float* Wc   = ws + Npad + 64;
  _Float16* W1t = (_Float16*)(ws + Npad + 256);              // 128*256 f16 = 16384 floats
  _Float16* W2t = (_Float16*)(ws + Npad + 256 + 16384);      //  64*128 f16 =  4096 floats
  _Float16* d1t = (_Float16*)(ws + Npad + 256 + 20480);      // 128*256 f16 = 16384 floats
  float* R1   = ws + Npad + 256 + 40960;
  float* R2   = R1 + (size_t)Nn * 128;
  float* feat = R1;                         // reused after both encodes
  float* x2a  = R2;                         // reused after z1 is dead

  float* out    = (float*)d_out;
  float* o_log  = out;                                        // [B,1]
  float* o_ret  = out + B;                                    // [N,2]
  float* o_x2o  = out + B + 2 * (size_t)Nn;                   // [N,64]
  float* o_log1 = out + B + 2 * (size_t)Nn + 64 * (size_t)Nn; // [B,128]

  const int* e_src = eidx;
  const int* e_dst = eidx + E;
  const int* idx0  = idx;
  const int* idx1  = idx + B;

  // pre-transpose weights to col-major f16
  k_wt<FIN, HID1><<<(FIN * HID1 + 255) / 256, 256, 0, stream>>>(W1, W1t);
  k_wt<HID1, HID2><<<(HID1 * HID2 + 255) / 256, 256, 0, stream>>>(W2, W2t);
  k_wt<256, 128><<<(256 * 128 + 255) / 256, 256, 0, stream>>>(dec1_W, d1t);

  // degree (self-loops give the initial 1.0), then dinv = rsqrt(deg)
  k_fill<<<(Nn + 255) / 256, 256, 0, stream>>>(dinv, 1.0f, Nn);
  k_deg_edges<<<(E + 255) / 256, 256, 0, stream>>>(e_dst, dinv, E);
  k_rsqrt_inplace<<<(Nn + 255) / 256, 256, 0, stream>>>(dinv, Nn);

  for (int which = 0; which < 2; ++which) {
    const float* X = which ? x_a : x_o;
    float* x2dst   = which ? x2a : o_x2o;
    float* h1   = R1;                        // [N,128]
    float* z1   = R2;                        // [N,128] (agg1, then PReLU'd in place)
    float* h2   = R1;                        // [N,64]  (R1 free after layer-1 post)
    float* agg2 = R1 + (size_t)Nn * 64;      // [N,64]

    // layer 1: h1 = X @ W1  (WMMA), scatter-add, self-loop + bias + PReLU
    int waves1 = Nn / 16;
    k_gemm_wmma<FIN, HID1><<<(waves1 + 3) / 4, 128, 0, stream>>>(X, W1t, h1, Nn);
    k_fill<<<(Nn * 128 + 255) / 256, 256, 0, stream>>>(z1, 0.f, Nn * 128);
    k_agg_edges<128><<<(E + 7) / 8, 256, 0, stream>>>(e_src, e_dst, dinv, h1, z1, E);
    k_layer_post<128><<<(Nn * 128 + 255) / 256, 256, 0, stream>>>(z1, h1, dinv, b1, a1, z1, Nn);

    // layer 2: h2 = z1 @ W2 (WMMA), scatter-add, self-loop + bias + PReLU -> x2
    k_gemm_wmma<HID1, HID2><<<(waves1 + 3) / 4, 128, 0, stream>>>(z1, W2t, h2, Nn);
    k_fill<<<(Nn * 64 + 255) / 256, 256, 0, stream>>>(agg2, 0.f, Nn * 64);
    k_agg_edges<64><<<(E + 7) / 8, 256, 0, stream>>>(e_src, e_dst, dinv, h2, agg2, E);
    k_layer_post<64><<<(Nn * 64 + 255) / 256, 256, 0, stream>>>(agg2, h2, dinv, b2, a2, x2dst, Nn);
  }

  // summary vector -> Wc = disc_W @ (sigmoid(mean(x2_o)) @ mlp1_W + mlp1_b)
  k_fill<<<1, 64, 0, stream>>>(csum, 0.f, 64);
  k_colsum<<<(Nn + 511) / 512, 64, 0, stream>>>(o_x2o, csum, Nn);
  k_summary<<<1, 64, 0, stream>>>(csum, mlp1_W, mlp1_b, disc_W, Wc);

  // bilinear discriminator scores -> ret_os [N,2]
  k_disc<<<(Nn + 255) / 256, 256, 0, stream>>>(o_x2o, x2a, Wc, disc_b, o_ret, Nn);

  // pair decoder: feat [B,256] -> log1 = relu(feat@dec1_W + b) -> log = log1@dec2_W + b
  k_feat<<<(B * 64 + 255) / 256, 256, 0, stream>>>(o_x2o, idx0, idx1, feat, B);
  int wavesd = B / 16;
  k_gemm_wmma<256, 128><<<(wavesd + 3) / 4, 128, 0, stream>>>(feat, d1t, o_log1, B);
  k_bias_relu<<<(B * 128 + 255) / 256, 256, 0, stream>>>(o_log1, dec1_b, B * 128, 128);
  k_dec2<<<(B + 255) / 256, 256, 0, stream>>>(o_log1, dec2_W, dec2_b, o_log, B);
}
